// GFConvNet_33569464385815
// MI455X (gfx1250) — 8-bit matrix operand order repaired and checked by reading, compiled, not run on hardware
//
#include <hip/hip_runtime.h>
#include <cstdint>

#define EPSV 1e-5f

typedef int v8i __attribute__((ext_vector_type(8)));

__global__ void k_binarize(const float* __restrict__ x, int8_t* __restrict__ o, int n) {
    int i = blockIdx.x * 256 + threadIdx.x;
    if (i < n) o[i] = (x[i] >= 0.f) ? (int8_t)1 : (int8_t)-1;
}

__global__ void k_conv1(const float* __restrict__ x, const int8_t* __restrict__ w,
                        float* __restrict__ out) {
    int i = blockIdx.x * 256 + threadIdx.x;
    int px = i & 31;
    int py = (i >> 5) & 31;
    int co = (i >> 10) & 127;
    int b  = i >> 17;
    const float* xb = x + (size_t)b * 3 * 1024;
    const int8_t* wr = w + co * 27;
    float s = 0.f;
    for (int ci = 0; ci < 3; ++ci)
        for (int kh = 0; kh < 3; ++kh) {
            int iy = py + kh - 1;
            if ((unsigned)iy >= 32u) continue;
            for (int kw = 0; kw < 3; ++kw) {
                int ix = px + kw - 1;
                if ((unsigned)ix >= 32u) continue;
                s += xb[(ci * 32 + iy) * 32 + ix] * (float)wr[(ci * 3 + kh) * 3 + kw];
            }
        }
    out[i] = s;
}

__global__ void __launch_bounds__(128)
k_conv_wmma(const int8_t* __restrict__ act, const int8_t* __restrict__ wgt,
            float* __restrict__ out, int Cin, int H, int W, int Cout) {
    const int K  = Cin * 9;
    const int HW = H * W;
    const int bI = blockIdx.z;
    const int tid  = threadIdx.x;
    const int lane = tid & 31;
    const int wv   = tid >> 5;
    const int h    = (lane >> 4) & 1;
    const int rn   = lane & 15;
    const int coutBase = (blockIdx.x << 6) + (wv << 4);
    const int posBase  = blockIdx.y << 6;

    __shared__ int8_t patch[64 * 72];

    const int pos_l = tid >> 1;
    const int half  = tid & 1;
    const int p  = posBase + pos_l;
    const int py = p / W;
    const int px = p - py * W;
    const int8_t* actB = act + (size_t)bI * Cin * HW;

    const int* arow = (const int*)(wgt + (size_t)(coutBase + rn) * K);

    v8i acc[4] = {};
    const int nChunks = K >> 6;
    for (int kc = 0; kc < nChunks; ++kc) {
        __syncthreads();
        const int kb = kc << 6;
        {
            int k0  = kb + (half << 5);
            int ci  = k0 / 9;
            int rem = k0 - ci * 9;
            int kh  = rem / 3;
            int kw  = rem - kh * 3;
            const int8_t* cbase = actB + (size_t)ci * HW;
            int8_t* dst = patch + pos_l * 72 + (half << 5);
            for (int j = 0; j < 32; ++j) {
                int iy = py + kh - 1;
                int ix = px + kw - 1;
                int8_t val = 0;
                if ((unsigned)iy < (unsigned)H && (unsigned)ix < (unsigned)W)
                    val = cbase[iy * W + ix];
                dst[j] = val;
                if (++kw == 3) { kw = 0; if (++kh == 3) { kh = 0; cbase += HW; } }
            }
        }
        __syncthreads();
        const int abase = kb >> 2;
        if (kc + 1 < nChunks)
            __builtin_prefetch(arow + abase + 16, 0, 0);
        v8i af;
        #pragma unroll
        for (int v = 0; v < 8; ++v)
            af[v] = arow[abase + ((v >> 1) << 2) + (v & 1) + (h << 1)];
        #pragma unroll
        for (int t = 0; t < 4; ++t) {
            const int8_t* prow = patch + ((t << 4) + rn) * 72;
            v8i bf;
            #pragma unroll
            for (int v = 0; v < 8; ++v)
                bf[v] = *(const int*)(prow + ((v >> 1) << 4) + ((v & 1) << 2) + (h << 3));
            acc[t] = __builtin_amdgcn_wmma_i32_16x16x64_iu8(true, af, true, bf,
                                                            acc[t], false, false);
        }
    }

    float* outB = out + ((size_t)bI * Cout + coutBase) * HW + posBase + rn;
    #pragma unroll
    for (int t = 0; t < 4; ++t) {
        #pragma unroll
        for (int v = 0; v < 8; ++v) {
            int m = v + (h << 3);
            outB[(size_t)m * HW + (t << 4)] = (float)acc[t][v];
        }
    }
}

__global__ void __launch_bounds__(128)
k_fc_wmma(const int8_t* __restrict__ act, const int8_t* __restrict__ wgt,
          float* __restrict__ out, int K, int Nout) {
    const int tid  = threadIdx.x;
    const int lane = tid & 31;
    const int wv   = tid >> 5;
    const int h    = (lane >> 4) & 1;
    const int rn   = lane & 15;
    const int bBase  = blockIdx.x << 4;
    const int oGroup = ((blockIdx.y << 2) + wv) << 6;

    const int* arow = (const int*)(act + (size_t)(bBase + rn) * K);
    const int* brow[4];
    #pragma unroll
    for (int t = 0; t < 4; ++t)
        brow[t] = (const int*)(wgt + (size_t)(oGroup + (t << 4) + rn) * K);

    v8i acc[4] = {};
    const int nChunks = K >> 6;
    for (int kc = 0; kc < nChunks; ++kc) {
        const int kb = kc << 4;
        v8i af;
        #pragma unroll
        for (int v = 0; v < 8; ++v)
            af[v] = arow[kb + ((v >> 1) << 2) + (v & 1) + (h << 1)];
        #pragma unroll
        for (int t = 0; t < 4; ++t) {
            v8i bf;
            #pragma unroll
            for (int v = 0; v < 8; ++v)
                bf[v] = brow[t][kb + ((v >> 1) << 2) + (v & 1) + (h << 1)];
            acc[t] = __builtin_amdgcn_wmma_i32_16x16x64_iu8(true, af, true, bf,
                                                            acc[t], false, false);
        }
    }
    #pragma unroll
    for (int t = 0; t < 4; ++t) {
        #pragma unroll
        for (int v = 0; v < 8; ++v) {
            int m = v + (h << 3);
            out[(size_t)(bBase + m) * Nout + oGroup + (t << 4) + rn] = (float)acc[t][v];
        }
    }
}

__global__ void k_pool(const float* __restrict__ in, float* __restrict__ out,
                       int C, int Hin, int Win, int total) {
    int i = blockIdx.x * 256 + threadIdx.x;
    if (i >= total) return;
    int Wo = Win >> 1, Ho = Hin >> 1;
    int wo = i % Wo;
    int ho = (i / Wo) % Ho;
    int c  = (i / (Wo * Ho)) % C;
    int b  = i / (Wo * Ho * C);
    const float* base = in + (((size_t)b * C + c) * Hin + 2 * ho) * Win + 2 * wo;
    float m0 = fmaxf(base[0], base[1]);
    float m1 = fmaxf(base[Win], base[Win + 1]);
    out[i] = fmaxf(m0, m1);
}

__global__ void k_bnstats(const float* __restrict__ x, float* __restrict__ mean,
                          float* __restrict__ var, int C, int B, int HW) {
    int c = blockIdx.x;
    int tid = threadIdx.x;
    int N = B * HW;
    float s = 0.f, s2 = 0.f;
    for (int i = tid; i < N; i += 256) {
        int b  = i / HW;
        int sp = i - b * HW;
        float v = x[((size_t)b * C + c) * HW + sp];
        s += v; s2 += v * v;
    }
    __shared__ float sh[256], sh2[256];
    sh[tid] = s; sh2[tid] = s2;
    __syncthreads();
    for (int st = 128; st > 0; st >>= 1) {
        if (tid < st) { sh[tid] += sh[tid + st]; sh2[tid] += sh2[tid + st]; }
        __syncthreads();
    }
    if (tid == 0) {
        float m = sh[0] / (float)N;
        mean[c] = m;
        var[c]  = sh2[0] / (float)N - m * m;
    }
}

__global__ void k_bn_bin(const float* __restrict__ x, const float* __restrict__ mean,
                         const float* __restrict__ var, const float* __restrict__ g,
                         const float* __restrict__ b, int8_t* __restrict__ out,
                         int C, int HW, int total) {
    int i = blockIdx.x * 256 + threadIdx.x;
    if (i >= total) return;
    int c = (i / HW) % C;
    float xh = (x[i] - mean[c]) * rsqrtf(var[c] + EPSV) * g[c] + b[c];
    out[i] = (xh >= 0.f) ? (int8_t)1 : (int8_t)-1;
}

__global__ void k_bn_out(const float* __restrict__ x, const float* __restrict__ mean,
                         const float* __restrict__ var, const float* __restrict__ g,
                         const float* __restrict__ b, float* __restrict__ out,
                         int C, int HW, int total) {
    int i = blockIdx.x * 256 + threadIdx.x;
    if (i >= total) return;
    int c = (i / HW) % C;
    out[i] = (x[i] - mean[c]) * rsqrtf(var[c] + EPSV) * g[c] + b[c];
}

__global__ void k_fc2(const int8_t* __restrict__ a, const int8_t* __restrict__ w,
                      const float* __restrict__ bias, float* __restrict__ out) {
    int i = blockIdx.x * 256 + threadIdx.x;
    if (i >= 256 * 10) return;
    int o = i % 10;
    int b = i / 10;
    const int8_t* ar = a + (size_t)b * 1024;
    const int8_t* wr = w + (size_t)o * 1024;
    int s = 0;
    for (int k = 0; k < 1024; ++k) s += (int)ar[k] * (int)wr[k];
    out[i] = (float)s + bias[o];
}

extern "C" void kernel_launch(void* const* d_in, const int* in_sizes, int n_in,
                              void* d_out, int out_size, void* d_ws, size_t ws_size,
                              hipStream_t stream) {
    (void)in_sizes; (void)n_in; (void)out_size; (void)ws_size;
    const float* x = (const float*)d_in[0];
    const float *w[6], *g[6], *bb[6];
    for (int i = 0; i < 6; ++i) {
        w[i]  = (const float*)d_in[1 + 3 * i];
        g[i]  = (const float*)d_in[2 + 3 * i];
        bb[i] = (const float*)d_in[3 + 3 * i];
    }
    const float* wf1   = (const float*)d_in[19];
    const float* gf1   = (const float*)d_in[20];
    const float* bf1   = (const float*)d_in[21];
    const float* wf2   = (const float*)d_in[22];
    const float* bias2 = (const float*)d_in[23];
    const float* gf2   = (const float*)d_in[24];
    const float* bf2   = (const float*)d_in[25];

    char* ws = (char*)d_ws;
    size_t off = 0;
    auto take = [&](size_t bytes) -> char* {
        char* p = ws + off;
        off += (bytes + 255) & ~(size_t)255;
        return p;
    };
    const int B = 256;
    float*  F0 = (float*)take((size_t)B * 128 * 32 * 32 * 4);
    float*  F1 = (float*)take((size_t)B * 128 * 16 * 16 * 4);
    int8_t* A0 = (int8_t*)take((size_t)B * 128 * 32 * 32);
    int8_t* A1 = (int8_t*)take((size_t)B * 128 * 32 * 32);
    int wsz[6] = {128 * 27, 128 * 128 * 9, 256 * 128 * 9,
                  256 * 256 * 9, 512 * 256 * 9, 512 * 512 * 9};
    int8_t* wb[6];
    for (int i = 0; i < 6; ++i) wb[i] = (int8_t*)take((size_t)wsz[i]);
    int8_t* wf1b = (int8_t*)take((size_t)1024 * 8192);
    int8_t* wf2b = (int8_t*)take((size_t)10 * 1024);
    float*  mean = (float*)take(1024 * 4);
    float*  var  = (float*)take(1024 * 4);

    for (int i = 0; i < 6; ++i)
        k_binarize<<<(wsz[i] + 255) / 256, 256, 0, stream>>>(w[i], wb[i], wsz[i]);
    k_binarize<<<(1024 * 8192) / 256, 256, 0, stream>>>(wf1, wf1b, 1024 * 8192);
    k_binarize<<<(10 * 1024 + 255) / 256, 256, 0, stream>>>(wf2, wf2b, 10 * 1024);

    {
        int total = B * 128 * 1024;
        k_conv1<<<total / 256, 256, 0, stream>>>(x, wb[0], F0);
        k_bnstats<<<128, 256, 0, stream>>>(F0, mean, var, 128, B, 1024);
        k_bn_bin<<<total / 256, 256, 0, stream>>>(F0, mean, var, g[0], bb[0], A0, 128, 1024, total);
    }
    {
        k_conv_wmma<<<dim3(2, 16, B), 128, 0, stream>>>(A0, wb[1], F0, 128, 32, 32, 128);
        int tp = B * 128 * 256;
        k_pool<<<tp / 256, 256, 0, stream>>>(F0, F1, 128, 32, 32, tp);
        k_bnstats<<<128, 256, 0, stream>>>(F1, mean, var, 128, B, 256);
        k_bn_bin<<<tp / 256, 256, 0, stream>>>(F1, mean, var, g[1], bb[1], A1, 128, 256, tp);
    }
    {
        k_conv_wmma<<<dim3(4, 4, B), 128, 0, stream>>>(A1, wb[2], F0, 128, 16, 16, 256);
        int t = B * 256 * 256;
        k_bnstats<<<256, 256, 0, stream>>>(F0, mean, var, 256, B, 256);
        k_bn_bin<<<t / 256, 256, 0, stream>>>(F0, mean, var, g[2], bb[2], A0, 256, 256, t);
    }
    {
        k_conv_wmma<<<dim3(4, 4, B), 128, 0, stream>>>(A0, wb[3], F0, 256, 16, 16, 256);
        int tp = B * 256 * 64;
        k_pool<<<tp / 256, 256, 0, stream>>>(F0, F1, 256, 16, 16, tp);
        k_bnstats<<<256, 256, 0, stream>>>(F1, mean, var, 256, B, 64);
        k_bn_bin<<<tp / 256, 256, 0, stream>>>(F1, mean, var, g[3], bb[3], A1, 256, 64, tp);
    }
    {
        k_conv_wmma<<<dim3(8, 1, B), 128, 0, stream>>>(A1, wb[4], F0, 256, 8, 8, 512);
        int t = B * 512 * 64;
        k_bnstats<<<512, 256, 0, stream>>>(F0, mean, var, 512, B, 64);
        k_bn_bin<<<t / 256, 256, 0, stream>>>(F0, mean, var, g[4], bb[4], A0, 512, 64, t);
    }
    {
        k_conv_wmma<<<dim3(8, 1, B), 128, 0, stream>>>(A0, wb[5], F0, 512, 8, 8, 512);
        int tp = B * 512 * 16;
        k_pool<<<tp / 256, 256, 0, stream>>>(F0, F1, 512, 8, 8, tp);
        k_bnstats<<<512, 256, 0, stream>>>(F1, mean, var, 512, B, 16);
        k_bn_bin<<<tp / 256, 256, 0, stream>>>(F1, mean, var, g[5], bb[5], A1, 512, 16, tp);
    }
    {
        k_fc_wmma<<<dim3(16, 4), 128, 0, stream>>>(A1, wf1b, F0, 8192, 1024);
        int t = B * 1024;
        k_bnstats<<<1024, 256, 0, stream>>>(F0, mean, var, 1024, B, 1);
        k_bn_bin<<<t / 256, 256, 0, stream>>>(F0, mean, var, gf1, bf1, A0, 1024, 1, t);
    }
    k_fc2<<<(2560 + 255) / 256, 256, 0, stream>>>(A0, wf2b, bias2, F1);
    k_bnstats<<<10, 256, 0, stream>>>(F1, mean, var, 10, B, 1);
    k_bn_out<<<2560 / 256, 256, 0, stream>>>(F1, mean, var, gf2, bf2,
                                             (float*)d_out, 10, 1, 2560);
}
